// Adapter_55104430408055
// MI455X (gfx1250) — matrix operand order repaired and checked by reading, not compiled, not run on hardware
//
#include <hip/hip_runtime.h>
#include <hip/hip_bf16.h>
#include <hip/hip_fp16.h>

// Problem constants (match reference)
#define PM 12          // routers
#define PN 8           // adapters
#define PC 1024        // model dim
#define PD 64          // down dim
#define PB 32          // batch
#define PS 512         // seq

#define SROWS 128      // rows of S per block
#define KCH   128      // K-chunk for down GEMM staging
#define NCH   128      // N(col)-chunk for up GEMM staging

#define WTS 136        // LDS row stride (bf16 elems) for transposed Wd  [64][136]
#define ZST 72         // LDS row stride for Z [128][72] and Wut [128][72]

typedef __attribute__((ext_vector_type(16))) __bf16 v16bf;
typedef __attribute__((ext_vector_type(8)))  __bf16 v8bf;
typedef __attribute__((ext_vector_type(8)))  float  v8f;

static __device__ __forceinline__ __bf16 tobf(float f) { return (__bf16)f; }

static __device__ __forceinline__ v16bf cat8(v8bf lo, v8bf hi) {
    return __builtin_shufflevector(lo, hi, 0,1,2,3,4,5,6,7,8,9,10,11,12,13,14,15);
}

// Load a 16x32 bf16 A-fragment for this lane directly from a global fp32 row.
// Layout: lane holds row (lane&15); lanes>=16 take
// the +16 K-half. Elements 0..15 = K khalf..khalf+15 (contiguous).
static __device__ __forceinline__ v16bf a_frag_from_global(const float* rowp, int kbase, int khalf) {
    const float4* p0 = (const float4*)(rowp + kbase + khalf);
    const float4* p1 = (const float4*)(rowp + kbase + khalf + 8);
    float4 f0 = p0[0], f1 = p0[1], f2 = p1[0], f3 = p1[1];
    v16bf a;
    a[0]=tobf(f0.x); a[1]=tobf(f0.y); a[2]=tobf(f0.z); a[3]=tobf(f0.w);
    a[4]=tobf(f1.x); a[5]=tobf(f1.y); a[6]=tobf(f1.z); a[7]=tobf(f1.w);
    a[8]=tobf(f2.x); a[9]=tobf(f2.y); a[10]=tobf(f2.z); a[11]=tobf(f2.w);
    a[12]=tobf(f3.x); a[13]=tobf(f3.y); a[14]=tobf(f3.z); a[15]=tobf(f3.w);
    return a;
}

// Load 16 contiguous bf16 from LDS (two 16B ds_load_b128s).
static __device__ __forceinline__ v16bf frag16_lds(const __bf16* p) {
    v8bf lo = *(const v8bf*)(p);
    v8bf hi = *(const v8bf*)(p + 8);
    return cat8(lo, hi);
}

__global__ __launch_bounds__(256, 1)
void adapter_moe_kernel(const float* __restrict__ x,
                        const int*   __restrict__ eidx,
                        const float* __restrict__ dw,
                        const float* __restrict__ db,
                        const float* __restrict__ uw,
                        float*       __restrict__ out) {
    __shared__ __bf16 s_wt[PD * WTS];      // transposed down weights chunk [d][k]
    __shared__ __bf16 s_z [SROWS * ZST];   // swish(down) activations, bf16
    __shared__ __bf16 s_wu[NCH * ZST];     // transposed up weights chunk [c][d]

    const int bid = blockIdx.x;
    const int m  = bid / (PB * (PS / SROWS));
    const int r0 = bid % (PB * (PS / SROWS));
    const int b  = r0 / (PS / SROWS);
    const int s0 = (r0 % (PS / SROWS)) * SROWS;

    const int e = eidx[m * PB + b];

    const float* wdbase = dw + ((size_t)(m * PN + e) * PC) * PD;   // [C][D]
    const float* wubase = uw + ((size_t)(m * PN + e) * PD) * PC;   // [D][C]
    const float* bias   = db + (size_t)(m * PN + e) * PD;
    const float* xbase  = x  + ((size_t)b * PS + s0) * PC;
    float*       obase  = out + (((size_t)(m * PB + b) * PS) + s0) * PC;

    const int tid  = threadIdx.x;
    const int wave = tid >> 5;
    const int lane = tid & 31;
    const int lrow  = lane & 15;       // row/col within tile
    const int lhi   = lane >> 4;       // K-half selector
    const int khalf = lhi * 16;        // A-frag K offset
    const int kb16  = lhi * 16;        // B-frag K offset
    const int wrow  = wave * 16;       // this wave's 16-row strip

    const float* xrow = xbase + (size_t)(wrow + lrow) * PC;

    // ---------------- Phase 1: Z = swish(X @ Wd + bias) ----------------
    v8f acc[4];
    #pragma unroll
    for (int t = 0; t < 4; ++t) acc[t] = (v8f){};

    #pragma unroll 1
    for (int kc = 0; kc < PC / KCH; ++kc) {
        __syncthreads();   // previous chunk fully consumed
        // Stage Wd[kc*128 .. +127][0..63] -> s_wt transposed, fp32 -> bf16
        const float* src = wdbase + (size_t)kc * KCH * PD;
        #pragma unroll
        for (int i = 0; i < 8; ++i) {
            int q  = tid + i * 256;          // float4 id over [128][16]
            int cl = q >> 4;                 // local k row
            int d0 = (q & 15) * 4;           // d start
            float4 f = *(const float4*)(src + cl * PD + d0);
            s_wt[(d0 + 0) * WTS + cl] = tobf(f.x);
            s_wt[(d0 + 1) * WTS + cl] = tobf(f.y);
            s_wt[(d0 + 2) * WTS + cl] = tobf(f.z);
            s_wt[(d0 + 3) * WTS + cl] = tobf(f.w);
        }
        // Prefetch next weight chunk into L2 (global_prefetch_b8)
        if (kc + 1 < PC / KCH)
            __builtin_prefetch(wdbase + (size_t)(kc + 1) * KCH * PD + tid * 32, 0, 0);
        __syncthreads();

        #pragma unroll
        for (int ks = 0; ks < KCH / 32; ++ks) {
            v16bf a = a_frag_from_global(xrow, kc * KCH + ks * 32, khalf);
            #pragma unroll
            for (int t = 0; t < 4; ++t) {
                int d = t * 16 + lrow;
                v16bf bfrag = frag16_lds(&s_wt[d * WTS + ks * 32 + kb16]);
                acc[t] = __builtin_amdgcn_wmma_f32_16x16x32_bf16(
                    false, a, false, bfrag, (short)0, acc[t], false, false);
            }
        }
    }

    // bias + swish, write Z (bf16) to LDS. C/D layout: elem v -> row v+8*lhi, col lrow.
    __syncthreads();
    #pragma unroll
    for (int t = 0; t < 4; ++t) {
        int d = t * 16 + lrow;
        float bv = bias[d];
        #pragma unroll
        for (int v = 0; v < 8; ++v) {
            int row = wrow + v + 8 * lhi;
            float z = acc[t][v] + bv;
            z = z / (1.0f + __expf(-z));      // swish
            s_z[row * ZST + d] = tobf(z);
        }
    }

    // ---------------- Phase 2: U = Z @ Wu ----------------
    #pragma unroll 1
    for (int nc = 0; nc < PC / NCH; ++nc) {
        __syncthreads();   // Z ready / previous Wu chunk consumed
        // Stage Wu[0..63][nc*128 .. +127] -> s_wu transposed [c][d]
        #pragma unroll
        for (int i = 0; i < 8; ++i) {
            int q  = tid + i * 256;          // float4 id over [64][32]
            int d  = q >> 5;
            int c4 = (q & 31) * 4;
            float4 f = *(const float4*)(wubase + (size_t)d * PC + nc * NCH + c4);
            s_wu[(c4 + 0) * ZST + d] = tobf(f.x);
            s_wu[(c4 + 1) * ZST + d] = tobf(f.y);
            s_wu[(c4 + 2) * ZST + d] = tobf(f.z);
            s_wu[(c4 + 3) * ZST + d] = tobf(f.w);
        }
        if (nc + 1 < PC / NCH)
            __builtin_prefetch(wubase + nc * NCH + NCH + (size_t)(tid >> 2) * PC + (tid & 3) * 32, 0, 0);
        __syncthreads();

        v8f uacc[8];
        #pragma unroll
        for (int n = 0; n < 8; ++n) uacc[n] = (v8f){};

        #pragma unroll
        for (int ks = 0; ks < PD / 32; ++ks) {
            int k0 = ks * 32;
            // A-frag from Z in LDS: two 16B chunks per lane
            v8bf lo = *(const v8bf*)(&s_z[(wrow + lrow) * ZST + k0 + khalf]);
            v8bf hi = *(const v8bf*)(&s_z[(wrow + lrow) * ZST + k0 + khalf + 8]);
            v16bf a = cat8(lo, hi);
            #pragma unroll
            for (int n = 0; n < 8; ++n) {
                int cl = n * 16 + lrow;
                v16bf bfrag = frag16_lds(&s_wu[cl * ZST + k0 + kb16]);
                uacc[n] = __builtin_amdgcn_wmma_f32_16x16x32_bf16(
                    false, a, false, bfrag, (short)0, uacc[n], false, false);
            }
        }

        // Streaming (non-temporal) stores of the 805MB output
        #pragma unroll
        for (int n = 0; n < 8; ++n) {
            int col = nc * NCH + n * 16 + lrow;
            #pragma unroll
            for (int v = 0; v < 8; ++v) {
                int row = wrow + v + 8 * lhi;
                __builtin_nontemporal_store(uacc[n][v], obase + (size_t)row * PC + col);
            }
        }
    }
}

extern "C" void kernel_launch(void* const* d_in, const int* in_sizes, int n_in,
                              void* d_out, int out_size, void* d_ws, size_t ws_size,
                              hipStream_t stream) {
    const float* x    = (const float*)d_in[0];
    const int*   eidx = (const int*)  d_in[1];
    const float* dw   = (const float*)d_in[2];
    const float* db   = (const float*)d_in[3];
    const float* uw   = (const float*)d_in[4];
    float*       out  = (float*)d_out;

    dim3 grid(PM * PB * (PS / SROWS));   // 1536 blocks
    dim3 block(256);                     // 8 wave32s
    adapter_moe_kernel<<<grid, block, 0, stream>>>(x, eidx, dw, db, uw, out);
}
